// FreMixer_77953656422926
// MI455X (gfx1250) — matrix operand order repaired and checked by reading, not compiled, not run on hardware
//
#include <hip/hip_runtime.h>

// ---------------------------------------------------------------------------
// FreMixer forward for MI455X (gfx1250, wave32).
// Dominant op: flatten einsum = batched GEMM 256x8192 @ 8192x4096 (x2 batches)
//   -> bf16 staging (134MB, L2-resident) + V_WMMA_F32_16X16X32_BF16 tiles.
//   GEMM wave tile: 32(c) x 64(m), software-pipelined (double-buffered frags).
// Everything else is small f32 elementwise/conv work.
// ---------------------------------------------------------------------------

#define B_   2
#define C_   256
#define N_   8192
#define H_   64
#define W_   64
#define M_   4096   // H_*W_
#define EPS_ 1e-5f

typedef __attribute__((ext_vector_type(16))) __bf16 v16bf;
typedef __attribute__((ext_vector_type(8)))  float  v8f;

struct alignas(16) U128 { unsigned int x, y, z, w; };
union UF { U128 q[2]; v16bf v; };
struct Frags { UF a0, a1, b0, b1, b2, b3; };

__device__ __forceinline__ unsigned short f2bf(float f) {
  unsigned u = __float_as_uint(f);
  u += 0x7FFFu + ((u >> 16) & 1u);      // round-to-nearest-even
  return (unsigned short)(u >> 16);
}

__device__ __forceinline__ float actf(float x, int a) {
  if (a == 1) return 0.5f * x * (1.f + erff(x * 0.70710678118654752f)); // exact gelu
  if (a == 2) return fmaxf(x, 0.f);
  if (a == 3) return tanhf(x);
  return x;
}

// ---------------- prep: flatten_mat f32 -> bf16 ----------------------------
__global__ void k_fm_to_bf16(const float* __restrict__ src,
                             unsigned short* __restrict__ dst, long n4) {
  long i = (long)blockIdx.x * blockDim.x + threadIdx.x;
  if (i >= n4) return;
  float4 v = ((const float4*)src)[i];
  uint2 o;
  o.x = (unsigned)f2bf(v.x) | ((unsigned)f2bf(v.y) << 16);
  o.y = (unsigned)f2bf(v.z) | ((unsigned)f2bf(v.w) << 16);
  ((uint2*)dst)[i] = o;
}

// ---------------- prep: eval-BN(tokens) -> bf16 ----------------------------
__global__ void k_bn_to_bf16(const float* __restrict__ tok,
                             const float* __restrict__ g, const float* __restrict__ be,
                             const float* __restrict__ mu, const float* __restrict__ va,
                             unsigned short* __restrict__ dst) {
  long i = (long)blockIdx.x * blockDim.x + threadIdx.x;   // float4 index
  long n4 = (long)B_ * C_ * N_ / 4;
  if (i >= n4) return;
  int c = (int)(((i * 4) / N_) % C_);
  float inv = g[c] * rsqrtf(va[c] + EPS_);
  float sh  = be[c] - mu[c] * inv;
  float4 v = ((const float4*)tok)[i];
  uint2 o;
  o.x = (unsigned)f2bf(v.x * inv + sh) | ((unsigned)f2bf(v.y * inv + sh) << 16);
  o.y = (unsigned)f2bf(v.z * inv + sh) | ((unsigned)f2bf(v.w * inv + sh) << 16);
  ((uint2*)dst)[i] = o;
}

// ---------------- big GEMM: out[b,c,m] = sum_n A[b,c,n]*Bm[b,m,n] ----------
// wave32, 8 waves/block arranged 2(c) x 4(m); wave tile = 32(c) x 64(m).
__device__ __forceinline__ void ldfr(Frags& f, const unsigned short* ap0,
                                     const unsigned short* ap1,
                                     const unsigned short* bp, int k) {
  f.a0.q[0] = *(const U128*)(ap0 + k);       f.a0.q[1] = *(const U128*)(ap0 + k + 8);
  f.a1.q[0] = *(const U128*)(ap1 + k);       f.a1.q[1] = *(const U128*)(ap1 + k + 8);
  const unsigned short* p = bp + k;
  f.b0.q[0] = *(const U128*)(p);             f.b0.q[1] = *(const U128*)(p + 8);
  f.b1.q[0] = *(const U128*)(p + 16ll * N_); f.b1.q[1] = *(const U128*)(p + 16ll * N_ + 8);
  f.b2.q[0] = *(const U128*)(p + 32ll * N_); f.b2.q[1] = *(const U128*)(p + 32ll * N_ + 8);
  f.b3.q[0] = *(const U128*)(p + 48ll * N_); f.b3.q[1] = *(const U128*)(p + 48ll * N_ + 8);
}

#define WMMA_BF16(A_, B_f, C_f) \
  __builtin_amdgcn_wmma_f32_16x16x32_bf16(false, (A_).v, false, (B_f).v, (short)0, (C_f), false, false)

__global__ __launch_bounds__(256) void k_flatten_gemm(
    const unsigned short* __restrict__ A,   // (B,256,8192) bf16
    const unsigned short* __restrict__ Bm,  // (B,4096,8192) bf16
    float* __restrict__ out)                // (B,256,4096) f32
{
  const int lane = threadIdx.x & 31;
  const int wave = threadIdx.x >> 5;
  const int wc = wave >> 2, wm = wave & 3;
  const int b  = blockIdx.z;
  const int c0 = blockIdx.y * 64 + wc * 32;
  const int m0 = blockIdx.x * 256 + wm * 64;
  const int r16 = lane & 15;
  const int hi  = lane >> 4;

  // A frag: lane = row M=c; lanes 0-15 hold K [0,16), lanes 16-31 K [16,32)
  const unsigned short* ap0 = A + ((size_t)b * C_ + (size_t)(c0 + r16)) * N_ + hi * 16;
  const unsigned short* ap1 = ap0 + 16ll * N_;
  // B frag: lane = col N=m; lanes 0-15 hold K [0,16), lanes 16-31 K [16,32)
  const unsigned short* bp = Bm + ((size_t)b * M_ + (size_t)(m0 + r16)) * N_ + hi * 16;

  v8f a00 = {}, a01 = {}, a02 = {}, a03 = {};
  v8f a10 = {}, a11 = {}, a12 = {}, a13 = {};

  Frags f0, f1;
  ldfr(f0, ap0, ap1, bp, 0);
  for (int k = 0; k < N_; k += 64) {
    // issue next-step loads before consuming current fragments
    ldfr(f1, ap0, ap1, bp, k + 32);
    a00 = WMMA_BF16(f0.a0, f0.b0, a00);
    a01 = WMMA_BF16(f0.a0, f0.b1, a01);
    a02 = WMMA_BF16(f0.a0, f0.b2, a02);
    a03 = WMMA_BF16(f0.a0, f0.b3, a03);
    a10 = WMMA_BF16(f0.a1, f0.b0, a10);
    a11 = WMMA_BF16(f0.a1, f0.b1, a11);
    a12 = WMMA_BF16(f0.a1, f0.b2, a12);
    a13 = WMMA_BF16(f0.a1, f0.b3, a13);
    int kn = k + 64;
    if (kn >= N_) kn = 0;                 // harmless wrap keeps the loop branch-free
    ldfr(f0, ap0, ap1, bp, kn);
    a00 = WMMA_BF16(f1.a0, f1.b0, a00);
    a01 = WMMA_BF16(f1.a0, f1.b1, a01);
    a02 = WMMA_BF16(f1.a0, f1.b2, a02);
    a03 = WMMA_BF16(f1.a0, f1.b3, a03);
    a10 = WMMA_BF16(f1.a1, f1.b0, a10);
    a11 = WMMA_BF16(f1.a1, f1.b1, a11);
    a12 = WMMA_BF16(f1.a1, f1.b2, a12);
    a13 = WMMA_BF16(f1.a1, f1.b3, a13);
  }

  // C/D layout: N = lane&15 ; M = vgpr + 8*(lane>=16)
  float* op0 = out + ((size_t)b * C_ + (size_t)(c0 + hi * 8)) * M_ + m0 + r16;
  float* op1 = op0 + 16ll * M_;
#pragma unroll
  for (int r = 0; r < 8; ++r) {
    op0[(size_t)r * M_ +  0] = a00[r];
    op0[(size_t)r * M_ + 16] = a01[r];
    op0[(size_t)r * M_ + 32] = a02[r];
    op0[(size_t)r * M_ + 48] = a03[r];
    op1[(size_t)r * M_ +  0] = a10[r];
    op1[(size_t)r * M_ + 16] = a11[r];
    op1[(size_t)r * M_ + 32] = a12[r];
    op1[(size_t)r * M_ + 48] = a13[r];
  }
}

// ---------------- generic 1x1 conv: out[b,o,p] = act(bias+W.in) ------------
__global__ void k_conv1x1(const float* __restrict__ in, const float* __restrict__ w,
                          const float* __restrict__ bias, float* __restrict__ out,
                          int Cin, int Cout, int P, int act, int octot, int ocofs) {
  long i = (long)blockIdx.x * blockDim.x + threadIdx.x;
  long tot = (long)B_ * Cout * P;
  if (i >= tot) return;
  int p = (int)(i % P);
  int o = (int)((i / P) % Cout);
  int b = (int)(i / ((long)P * Cout));
  const float* ip = in + (size_t)b * Cin * P + p;
  const float* wp = w + (size_t)o * Cin;
  float acc = bias[o];
#pragma unroll 4
  for (int k = 0; k < Cin; ++k) acc += wp[k] * ip[(size_t)k * P];
  out[((size_t)b * octot + ocofs + o) * P + p] = actf(acc, act);
}

// ------- lifting: reflect-pad(2,1) k=4 conv (64->128 ch) + ReLU ------------
// src is a strided view; axis: 0 = taps along x, 1 = taps along y.
__global__ void k_lift_conv4(const float* __restrict__ src, long sb, long sc, long sy, long sx,
                             int Hs, int Ws, const float* __restrict__ w1,
                             const float* __restrict__ b1, float* __restrict__ h, int axis) {
  long i = (long)blockIdx.x * blockDim.x + threadIdx.x;
  long tot = (long)B_ * 128 * Hs * Ws;
  if (i >= tot) return;
  int x = (int)(i % Ws);
  int y = (int)((i / Ws) % Hs);
  int o = (int)((i / ((long)Ws * Hs)) % 128);
  int b = (int)(i / ((long)Ws * Hs * 128));
  int L  = axis ? Hs : Ws;
  int c0 = axis ? y : x;
  long tap[4];
#pragma unroll
  for (int t = 0; t < 4; ++t) {
    int p = c0 + t - 2;
    if (p < 0) p = -p;                   // reflect (no edge repeat)
    if (p >= L) p = 2 * L - 2 - p;
    tap[t] = axis ? ((long)p * sy + (long)x * sx) : ((long)y * sy + (long)p * sx);
  }
  const float* sp = src + (long)b * sb;
  const float* wp = w1 + (size_t)o * 256;   // 64 ic * 4 taps
  float acc = b1[o];
  for (int ic = 0; ic < 64; ++ic) {
    const float* s2 = sp + (long)ic * sc;
    acc += wp[ic * 4 + 0] * s2[tap[0]] + wp[ic * 4 + 1] * s2[tap[1]]
         + wp[ic * 4 + 2] * s2[tap[2]] + wp[ic * 4 + 3] * s2[tap[3]];
  }
  h[i] = fmaxf(acc, 0.f);   // dense (B,128,Hs,Ws)
}

// ------- lifting: 1x1 (128->64) + tanh, combined with other +/- -----------
__global__ void k_lift_combine(const float* __restrict__ h, const float* __restrict__ w2,
                               const float* __restrict__ b2,
                               const float* __restrict__ oth, long ob, long oc, long oy, long ox,
                               float* __restrict__ out, int octot, int ocofs,
                               int Hs, int Ws, float sign) {
  long i = (long)blockIdx.x * blockDim.x + threadIdx.x;
  long tot = (long)B_ * 64 * Hs * Ws;
  if (i >= tot) return;
  int x = (int)(i % Ws);
  int y = (int)((i / Ws) % Hs);
  int c = (int)((i / ((long)Ws * Hs)) % 64);
  int b = (int)(i / ((long)Ws * Hs * 64));
  long sp = (long)Hs * Ws;
  const float* hp = h + (long)b * 128 * sp + (long)y * Ws + x;
  const float* wp = w2 + (size_t)c * 128;
  float acc = b2[c];
#pragma unroll 4
  for (int o = 0; o < 128; ++o) acc += wp[o] * hp[(long)o * sp];
  float ov = oth[(long)b * ob + (long)c * oc + (long)y * oy + (long)x * ox];
  out[((size_t)b * octot + ocofs + c) * sp + (long)y * Ws + x] = ov + sign * tanhf(acc);
}

// ------- ConvTranspose2d k=4 s=2 p=1, w:(Cin,Cout,4,4), fused eval-BN ------
__global__ void k_convt4x4(const float* __restrict__ in, const float* __restrict__ w,
                           const float* __restrict__ bias,
                           const float* __restrict__ g, const float* __restrict__ be,
                           const float* __restrict__ mu, const float* __restrict__ va,
                           float* __restrict__ out, int Cin, int Cout, int Hi, int Wi,
                           int octot, int ocofs) {
  int Ho = 2 * Hi, Wo = 2 * Wi;
  long i = (long)blockIdx.x * blockDim.x + threadIdx.x;
  long tot = (long)B_ * Cout * Ho * Wo;
  if (i >= tot) return;
  int ox = (int)(i % Wo);
  int oy = (int)((i / Wo) % Ho);
  int o  = (int)((i / ((long)Wo * Ho)) % Cout);
  int b  = (int)(i / ((long)Wo * Ho * Cout));
  float acc = bias[o];
  const float* ib = in + (size_t)b * Cin * Hi * Wi;
  long ws_ = (long)Cout * 16, is_ = (long)Hi * Wi;
  for (int ky = (oy + 1) & 1; ky < 4; ky += 2) {
    int iy = (oy + 1 - ky) >> 1;
    if (iy < 0 || iy >= Hi) continue;
    for (int kx = (ox + 1) & 1; kx < 4; kx += 2) {
      int ix = (ox + 1 - kx) >> 1;
      if (ix < 0 || ix >= Wi) continue;
      const float* ip = ib + (long)iy * Wi + ix;
      const float* wp = w + (size_t)o * 16 + ky * 4 + kx;
#pragma unroll 4
      for (int ci = 0; ci < Cin; ++ci) acc += wp[(long)ci * ws_] * ip[(long)ci * is_];
    }
  }
  float inv = g[o] * rsqrtf(va[o] + EPS_);
  acc = acc * inv + (be[o] - mu[o] * inv);
  out[((size_t)b * octot + ocofs + o) * ((long)Ho * Wo) + (long)oy * Wo + ox] = acc;
}

// ------- depthwise 3x3 pad=1; optional summed 2nd input / input relu / scale
__global__ void k_dw3x3(const float* __restrict__ a, const float* __restrict__ b2,
                        const float* __restrict__ w, const float* __restrict__ bias,
                        const float* __restrict__ scale, float* __restrict__ out,
                        int in_relu) {
  long i = (long)blockIdx.x * blockDim.x + threadIdx.x;
  long tot = (long)B_ * C_ * H_ * W_;
  if (i >= tot) return;
  int x = (int)(i % W_);
  int y = (int)((i / W_) % H_);
  int c = (int)((i / (W_ * H_)) % C_);
  long base = i - x - (long)y * W_;
  float acc = bias[c];
  const float* wc = w + c * 9;
#pragma unroll
  for (int dy = 0; dy < 3; ++dy) {
    int yy = y + dy - 1;
    if (yy < 0 || yy >= H_) continue;
#pragma unroll
    for (int dx = 0; dx < 3; ++dx) {
      int xx = x + dx - 1;
      if (xx < 0 || xx >= W_) continue;
      long idx = base + (long)yy * W_ + xx;
      float v = a[idx];
      if (b2) v += b2[idx];
      if (in_relu) v = fmaxf(v, 0.f);
      acc += wc[dy * 3 + dx] * v;
    }
  }
  if (scale) acc *= scale[c];
  out[i] = acc;
}

// ------- final gather + residual -------------------------------------------
__global__ void k_gather_out(const float* __restrict__ tokens, const int* __restrict__ idx,
                             const float* __restrict__ g, float* __restrict__ out) {
  long i = (long)blockIdx.x * blockDim.x + threadIdx.x;
  long tot = (long)B_ * C_ * N_;
  if (i >= tot) return;
  int bc = (int)(i / N_);
  int j = idx[i];
  out[i] = tokens[i] + g[(size_t)bc * M_ + j];
}

// ===========================================================================
struct HView { const float* p; long sb, sc, sy, sx; };

extern "C" void kernel_launch(void* const* d_in, const int* in_sizes, int n_in,
                              void* d_out, int out_size, void* d_ws, size_t ws_size,
                              hipStream_t stream) {
  (void)in_sizes; (void)n_in; (void)out_size; (void)ws_size;
  const float* tokens = (const float*)d_in[0];
  const float* flat   = (const float*)d_in[1];
  const int*   infl   = (const int*)d_in[2];
  const float* bng = (const float*)d_in[3], *bnb = (const float*)d_in[4];
  const float* bnm = (const float*)d_in[5], *bnv = (const float*)d_in[6];
  const float* red_w = (const float*)d_in[7], *red_b = (const float*)d_in[8];
  const float* lw1 = (const float*)d_in[9],  *lb1 = (const float*)d_in[10];
  const float* lw2 = (const float*)d_in[11], *lb2 = (const float*)d_in[12];
  const float* f2w1 = (const float*)d_in[13], *f2b1 = (const float*)d_in[14];
  const float* f2w2 = (const float*)d_in[15], *f2b2 = (const float*)d_in[16];
  const float* ct2w = (const float*)d_in[17], *ct2b = (const float*)d_in[18];
  const float* b2g = (const float*)d_in[19], *b2b = (const float*)d_in[20];
  const float* b2m = (const float*)d_in[21], *b2v = (const float*)d_in[22];
  const float* f1w1 = (const float*)d_in[23], *f1b1 = (const float*)d_in[24];
  const float* f1w2 = (const float*)d_in[25], *f1b2 = (const float*)d_in[26];
  const float* ct1w = (const float*)d_in[27], *ct1b = (const float*)d_in[28];
  const float* b1g = (const float*)d_in[29], *b1b = (const float*)d_in[30];
  const float* b1m = (const float*)d_in[31], *b1v = (const float*)d_in[32];
  const float* dw1w = (const float*)d_in[33], *dw1b = (const float*)d_in[34];
  const float* dw2w = (const float*)d_in[35], *dw2b = (const float*)d_in[36];
  const float* scw  = (const float*)d_in[37];

  // ---- workspace carve-out ----
  char* ws = (char*)d_ws;
  size_t off = 0;
  auto alloc = [&](size_t bytes) { void* p = ws + off; off += (bytes + 255) & ~(size_t)255; return p; };
  unsigned short* FM   = (unsigned short*)alloc((size_t)B_ * M_ * N_ * 2);   // 134 MB bf16
  unsigned short* RBN  = (unsigned short*)alloc((size_t)B_ * C_ * N_ * 2);   // 8.4 MB bf16
  float* RGRID = (float*)alloc((size_t)B_ * C_ * M_ * 4);        // (B,256,64,64)
  float* RR    = (float*)alloc((size_t)B_ * 64 * M_ * 4);        // (B,64,64,64)
  float* HBUF  = (float*)alloc((size_t)B_ * 128 * 2048 * 4);     // lift hidden (max)
  float* C1    = (float*)alloc((size_t)B_ * 64 * 2048 * 4);      // (B,64,64,32)
  float* D1    = (float*)alloc((size_t)B_ * 64 * 2048 * 4);
  float* X1    = (float*)alloc((size_t)B_ * 384 * 1024 * 4);     // (B,384,32,32)
  float* C2    = (float*)alloc((size_t)B_ * 64 * 512 * 4);       // (B,64,32,16)
  float* D2    = (float*)alloc((size_t)B_ * 64 * 512 * 4);
  float* X2    = (float*)alloc((size_t)B_ * 256 * 256 * 4);      // (B,256,16,16)
  float* TFF   = (float*)alloc((size_t)B_ * 256 * 1024 * 4);     // ff hidden (max)
  float* Y2A   = (float*)alloc((size_t)B_ * 256 * 256 * 4);      // ff2 out
  float* Y1A   = (float*)alloc((size_t)B_ * 256 * 1024 * 4);     // ff1 out
  float* Y1    = (float*)alloc((size_t)B_ * 256 * 4096 * 4);     // convT1+bn out
  float* T1    = (float*)alloc((size_t)B_ * 256 * 4096 * 4);     // dw1 out
  float* G     = (float*)alloc((size_t)B_ * 256 * 4096 * 4);     // dw2 out (scaled)

#define L1D(kern, total, ...) \
  kern<<<dim3((unsigned)(((total) + 255) / 256)), dim3(256), 0, stream>>>(__VA_ARGS__)

  // ---- prep: bf16 staging ----
  L1D(k_fm_to_bf16, (long)B_ * M_ * N_ / 4, flat, FM, (long)B_ * M_ * N_ / 4);
  L1D(k_bn_to_bf16, (long)B_ * C_ * N_ / 4, tokens, bng, bnb, bnm, bnv, RBN);

  // ---- dominant GEMM (WMMA bf16, pipelined) ----
  k_flatten_gemm<<<dim3(M_ / 256, C_ / 64, B_), dim3(256), 0, stream>>>(RBN, FM, RGRID);

  // ---- channel reduction 256->64 ----
  L1D(k_conv1x1, (long)B_ * 64 * M_, RGRID, red_w, red_b, RR, 256, 64, M_, 0, 64, 0);

  // ---- wavelet weights accessors ----
  auto w1p = [&](int l, int s, int n) { return lw1 + (size_t)((l * 2 + s) * 2 + n) * 128 * 64 * 4; };
  auto b1p = [&](int l, int s, int n) { return lb1 + (size_t)((l * 2 + s) * 2 + n) * 128; };
  auto w2p = [&](int l, int s, int n) { return lw2 + (size_t)((l * 2 + s) * 2 + n) * 64 * 128; };
  auto b2p = [&](int l, int s, int n) { return lb2 + (size_t)((l * 2 + s) * 2 + n) * 64; };

  // one lifting pass: d = odd - tanh(P(even)); c = even + tanh(U(d))
  auto lift = [&](HView even, HView odd, int Hs, int Ws, int axis,
                  const float* w1P, const float* b1P, const float* w2P, const float* b2P,
                  const float* w1U, const float* b1U, const float* w2U, const float* b2U,
                  float* dbuf, int dct, int dco, float* cbuf, int cct, int cco) {
    long tc = (long)B_ * 128 * Hs * Ws, tm = (long)B_ * 64 * Hs * Ws;
    L1D(k_lift_conv4, tc, even.p, even.sb, even.sc, even.sy, even.sx, Hs, Ws, w1P, b1P, HBUF, axis);
    L1D(k_lift_combine, tm, HBUF, w2P, b2P, odd.p, odd.sb, odd.sc, odd.sy, odd.sx,
        dbuf, dct, dco, Hs, Ws, -1.f);
    const float* dv = dbuf + (long)dco * Hs * Ws;
    L1D(k_lift_conv4, tc, dv, (long)dct * Hs * Ws, (long)Hs * Ws, (long)Ws, 1L, Hs, Ws, w1U, b1U, HBUF, axis);
    L1D(k_lift_combine, tm, HBUF, w2U, b2U, even.p, even.sb, even.sc, even.sy, even.sx,
        cbuf, cct, cco, Hs, Ws, 1.f);
  };

  // ---- wavelet level 1 on RR (B,64,64,64) ----
  lift({RR, 64L * 4096, 4096, 64, 2}, {RR + 1, 64L * 4096, 4096, 64, 2}, 64, 32, 0,
       w1p(0,0,0), b1p(0,0,0), w2p(0,0,0), b2p(0,0,0),
       w1p(0,0,1), b1p(0,0,1), w2p(0,0,1), b2p(0,0,1),
       D1, 64, 0, C1, 64, 0);
  // vertical on C1 -> LH1 (X1 ch64), LL1 (X1 ch0)
  lift({C1, 64L * 2048, 2048, 64, 1}, {C1 + 32, 64L * 2048, 2048, 64, 1}, 32, 32, 1,
       w1p(0,1,0), b1p(0,1,0), w2p(0,1,0), b2p(0,1,0),
       w1p(0,1,1), b1p(0,1,1), w2p(0,1,1), b2p(0,1,1),
       X1, 384, 64, X1, 384, 0);
  // vertical on D1 -> HH1 (X1 ch192), HL1 (X1 ch128)
  lift({D1, 64L * 2048, 2048, 64, 1}, {D1 + 32, 64L * 2048, 2048, 64, 1}, 32, 32, 1,
       w1p(0,1,0), b1p(0,1,0), w2p(0,1,0), b2p(0,1,0),
       w1p(0,1,1), b1p(0,1,1), w2p(0,1,1), b2p(0,1,1),
       X1, 384, 192, X1, 384, 128);

  // ---- wavelet level 2 on LL1 (view into X1) ----
  lift({X1, 384L * 1024, 1024, 32, 2}, {X1 + 1, 384L * 1024, 1024, 32, 2}, 32, 16, 0,
       w1p(1,0,0), b1p(1,0,0), w2p(1,0,0), b2p(1,0,0),
       w1p(1,0,1), b1p(1,0,1), w2p(1,0,1), b2p(1,0,1),
       D2, 64, 0, C2, 64, 0);
  // vertical on C2 -> LH2 (X2 ch64), LL2 (X2 ch0)
  lift({C2, 64L * 512, 512, 32, 1}, {C2 + 16, 64L * 512, 512, 32, 1}, 16, 16, 1,
       w1p(1,1,0), b1p(1,1,0), w2p(1,1,0), b2p(1,1,0),
       w1p(1,1,1), b1p(1,1,1), w2p(1,1,1), b2p(1,1,1),
       X2, 256, 64, X2, 256, 0);
  // vertical on D2 -> HH2 (X2 ch192), HL2 (X2 ch128)
  lift({D2, 64L * 512, 512, 32, 1}, {D2 + 16, 64L * 512, 512, 32, 1}, 16, 16, 1,
       w1p(1,1,0), b1p(1,1,0), w2p(1,1,0), b2p(1,1,0),
       w1p(1,1,1), b1p(1,1,1), w2p(1,1,1), b2p(1,1,1),
       X2, 256, 192, X2, 256, 128);

  // ---- ff2 (gelu MLP) + convT2 + bn2 -> X1 channels 256..383 ----
  L1D(k_conv1x1, (long)B_ * 256 * 256, X2, f2w1, f2b1, TFF, 256, 256, 256, 1, 256, 0);
  L1D(k_conv1x1, (long)B_ * 256 * 256, TFF, f2w2, f2b2, Y2A, 256, 256, 256, 0, 256, 0);
  L1D(k_convt4x4, (long)B_ * 128 * 1024, Y2A, ct2w, ct2b, b2g, b2b, b2m, b2v,
      X1, 256, 128, 16, 16, 384, 256);

  // ---- ff1 + convT1 + bn1 -> Y1 ----
  L1D(k_conv1x1, (long)B_ * 256 * 1024, X1, f1w1, f1b1, TFF, 384, 256, 1024, 1, 256, 0);
  L1D(k_conv1x1, (long)B_ * 256 * 1024, TFF, f1w2, f1b2, Y1A, 256, 256, 1024, 0, 256, 0);
  L1D(k_convt4x4, (long)B_ * 256 * 4096, Y1A, ct1w, ct1b, b1g, b1b, b1m, b1v,
      Y1, 256, 256, 32, 32, 256, 0);

  // ---- residual + depthwise stack + scale ----
  L1D(k_dw3x3, (long)B_ * C_ * M_, RGRID, Y1, dw1w, dw1b, (const float*)nullptr, T1, 0);
  L1D(k_dw3x3, (long)B_ * C_ * M_, T1, (const float*)nullptr, dw2w, dw2b, scw, G, 1);

  // ---- inflate gather + token residual ----
  L1D(k_gather_out, (long)B_ * C_ * N_, tokens, infl, G, (float*)d_out);

#undef L1D
}
